// ForceDime_45535243272624
// MI455X (gfx1250) — matrix operand order repaired and checked by reading, not compiled, not run on hardware
//
#include <hip/hip_runtime.h>
#include <hip/hip_bf16.h>
#include <math.h>

typedef __attribute__((ext_vector_type(16))) _Float16 v16h;
typedef __attribute__((ext_vector_type(8)))  _Float16 v8h;
typedef __attribute__((ext_vector_type(8)))  float    v8f;
typedef _Float16 h16;

#define DEV static __device__ __forceinline__

namespace {
constexpr int NA   = 20000;
constexpr int NE   = 300000;
constexpr int NANG = 1500000;
constexpr int NRBF = 6;
constexpr int LS   = 7;
constexpr int NS   = 6;
constexpr int NSBF = 42;   // LS*NS
constexpr float RCUT = 5.0f;
constexpr float EPS3 = 3.0f * 1e-15f;
constexpr float PI_F = 3.14159265358979f;
}

// zeros of spherical Bessel j_l, l=0..6, n=1..6 (standard tables)
__constant__ float cBES[LS][NS] = {
 { 3.14159265f,  6.28318531f,  9.42477796f, 12.56637061f, 15.70796327f, 18.84955592f},
 { 4.49340946f,  7.72525184f, 10.90412166f, 14.06619391f, 17.22075527f, 20.37130296f},
 { 5.76345920f,  9.09501133f, 12.32294097f, 15.51460301f, 18.68903636f, 21.85387422f},
 { 6.98793200f, 10.41711855f, 13.69802315f, 16.92362129f, 20.12180617f, 23.30424699f},
 { 8.18256145f, 11.70490715f, 15.03966471f, 18.30125596f, 21.52541773f, 24.72756555f},
 { 9.35581211f, 12.96653017f, 16.35470964f, 19.65315210f, 22.90455065f, 26.12775014f},
 {10.51283541f, 14.20739246f, 17.64797487f, 20.98346307f, 24.26276802f, 27.50786839f}};

DEV int   imin(int a, int b){ return a < b ? a : b; }
DEV float silu(float x){ return x * (1.0f / (1.0f + __expf(-x))); }
DEV void  wbar(){ __builtin_amdgcn_wave_barrier(); }

DEV float envelope6(float x){
  if(x >= 1.0f) return 0.0f;
  float x2 = x*x, x4 = x2*x2, x5 = x4*x;
  return 1.0f/x + x5*(-28.0f + x*(48.0f - 21.0f*x));
}

DEV float sphjl(int l, float x){
  float s, c; __sincosf(x, &s, &c);
  float inv = 1.0f / x;
  float j0 = s*inv;
  if(l == 0) return j0;
  float j1 = (s*inv - c)*inv;
  float jm = j0, jc = j1;
  for(int ll = 1; ll < l; ++ll){ float jn = (2.0f*ll+1.0f)*inv*jc - jm; jm = jc; jc = jn; }
  return jc;
}

// ---------------- WMMA fragment helpers (layouts per CDNA5 ISA) ----
// A 16x32 f16: lane L -> row L&15; kbase = ((L>>4)<<3); halves 0..7 -> K=kb+0..7,
// halves 8..15 -> K=kb+16..23.  Two contiguous 16B runs -> ds_load_b128 x2.
DEV v16h frag_a_lds(const h16* X, int ld, int kc, int lane){
  const h16* p = X + (lane & 15)*ld + kc + ((lane >> 4) << 3);
  v8h lo = *(const v8h*)p;
  v8h hi = *(const v8h*)(p + 16);
  return __builtin_shufflevector(lo, hi, 0,1,2,3,4,5,6,7,8,9,10,11,12,13,14,15);
}
// B fragments are pre-swizzled in LDS: Wz[((kc/32)*nT + n)*512 + lane*16 + h]
// holds W[kc + ((lane>>4)<<3) + (h&7) + ((h>>3)<<4)][n*16 + (lane&15)].  One lane reads 32
// contiguous bytes -> two ds_load_b128, zero packing VALU.
DEV v16h frag_b_swz(const h16* Wz, int tile, int lane){
  const h16* p = Wz + ((size_t)tile << 9) + lane*16;
  v8h lo = *(const v8h*)p;
  v8h hi = *(const v8h*)(p + 8);
  return __builtin_shufflevector(lo, hi, 0,1,2,3,4,5,6,7,8,9,10,11,12,13,14,15);
}
DEV v8f vzero(){ v8f z;
#pragma unroll
  for(int i = 0; i < 8; ++i) z[i] = 0.0f; return z; }

template<int NT>
DEV void gemm_swz(const h16* X, int xld, const h16* Wz, int K, v8f* acc, int lane){
  for(int kc = 0; kc < K; kc += 32){
    v16h a = frag_a_lds(X, xld, kc, lane);
#pragma unroll
    for(int n = 0; n < NT; ++n){
      v16h b = frag_b_swz(Wz, (kc >> 5)*NT + n, lane);
      acc[n] = __builtin_amdgcn_wmma_f32_16x16x32_f16(false, a, false, b, (short)0, acc[n], false, false);
    }
  }
}
// C/D f32: lane L, vgpr v -> row v + 8*(L>>4), col L&15.
template<int NT>
DEV void stage_acc_h16(h16* Y, int yld, const v8f* acc, const float* bias, bool do_act, int lane){
  const int colb = lane & 15, rb = (lane >> 4) * 8;
#pragma unroll
  for(int n = 0; n < NT; ++n)
#pragma unroll
    for(int v = 0; v < 8; ++v){
      float x = acc[n][v]; int c = n*16 + colb;
      if(bias) x += bias[c];
      if(do_act) x = silu(x);
      Y[(rb+v)*yld + c] = (h16)x;
    }
}
template<int NT>
DEV void store_acc_glb(float* G, int row0, int rmax, const v8f* acc, const float* bias, bool do_act, int lane){
  const int colb = lane & 15, rb = (lane >> 4) * 8;
#pragma unroll
  for(int n = 0; n < NT; ++n)
#pragma unroll
    for(int v = 0; v < 8; ++v){
      int r = row0 + rb + v;
      if(r < rmax){
        float x = acc[n][v]; int c = n*16 + colb;
        if(bias) x += bias[c];
        if(do_act) x = silu(x);
        G[(size_t)r*64 + c] = x;
      }
    }
}
// Pre-swizzle a KxN f32 weight into B-fragment order (f16).  Rows k>=Ksrc pad 0.
DEV void load_wswz(h16* dst, const float* src, int Ksrc, int K, int N, int tid, int nthr){
  const int nT = N >> 4;
  const int total = K * N;
  for(int i = tid; i < total; i += nthr){
    int h    = i & 15;
    int lane = (i >> 4) & 31;
    int tile = i >> 9;
    int kcI  = tile / nT;
    int n    = tile - kcI*nT;
    int k    = kcI*32 + ((lane >> 4) << 3) + (h & 7) + ((h >> 3) << 4);
    int col  = n*16 + (lane & 15);
    dst[i] = (h16)(k < Ksrc ? src[k*N + col] : 0.0f);
  }
}
DEV void load_f32(float* dst, const float* src, int n, int tid, int nthr){
  for(int i = tid; i < n; i += nthr) dst[i] = src[i];
}

// ---------------- simple kernels -------------------------------------------
__global__ void k_zero(float* p, long long n){
  long long i = (long long)blockIdx.x*blockDim.x + threadIdx.x;
  if(i < n) p[i] = 0.0f;
}

__global__ void k_edge_geom(const float* __restrict__ nxyz, const int* __restrict__ nbr,
                            float* __restrict__ d_e, float* __restrict__ erbf){
  int e = blockIdx.x*blockDim.x + threadIdx.x;
  if(e >= NE) return;
  int i0 = nbr[2*e], i1 = nbr[2*e+1];
  float dx = nxyz[4*i0+1]-nxyz[4*i1+1];
  float dy = nxyz[4*i0+2]-nxyz[4*i1+2];
  float dz = nxyz[4*i0+3]-nxyz[4*i1+3];
  float d = sqrtf(dx*dx + dy*dy + dz*dz);
  d_e[e] = d;
  float x = d * (1.0f/RCUT);
  float env = envelope6(x);
#pragma unroll
  for(int n = 0; n < NRBF; ++n) erbf[(size_t)e*NRBF + n] = env * __sinf(PI_F*(n+1)*x);
}

__global__ void k_angle_sbf(const float* __restrict__ nxyz, const int* __restrict__ ang,
                            const int* __restrict__ kj, const float* __restrict__ d_e,
                            float* __restrict__ asbf){
  int a = blockIdx.x*blockDim.x + threadIdx.x;
  if(a >= NANG) return;
  int i = ang[3*a], j = ang[3*a+1], k = ang[3*a+2];
  float jx = nxyz[4*j+1], jy = nxyz[4*j+2], jz = nxyz[4*j+3];
  float ax = nxyz[4*i+1]-jx, ay = nxyz[4*i+2]-jy, az = nxyz[4*i+3]-jz;
  float bx = nxyz[4*k+1]-jx, by = nxyz[4*k+2]-jy, bz = nxyz[4*k+3]-jz;
  float na = sqrtf(ax*ax+ay*ay+az*az + EPS3);
  float nb = sqrtf(bx*bx+by*by+bz*bz + EPS3);
  float cs = (ax*bx+ay*by+az*bz) / (na*nb);
  cs = fminf(fmaxf(cs, -1.0f+1e-7f), 1.0f-1e-7f);
  float x = d_e[kj[a]] * (1.0f/RCUT);
  float env = envelope6(x);
  float P[LS]; P[0] = 1.0f; P[1] = cs;
#pragma unroll
  for(int l = 2; l < LS; ++l) P[l] = ((2.0f*l-1.0f)*cs*P[l-1] - (l-1.0f)*P[l-2]) * (1.0f/l);
  float* row = asbf + (size_t)a*NSBF;
#pragma unroll
  for(int l = 0; l < LS; ++l){
    float ep = env * P[l];
#pragma unroll
    for(int n = 0; n < NS; ++n) row[l*NS + n] = ep * sphjl(l, x*cBES[l][n]);
  }
}

__global__ void k_embed_h(const float* __restrict__ nxyz, const float* __restrict__ table,
                          float* __restrict__ h){
  int i = blockIdx.x*blockDim.x + threadIdx.x;
  if(i >= NA*64) return;
  int a = i >> 6, c = i & 63;
  int z = (int)nxyz[4*a];
  h[i] = table[z*64 + c];
}

// ---------------- embedding m_ji = act(cat(h_j,h_i,e_emb) @ W + b) -----------
__global__ void k_embed_m(const float* __restrict__ h, const float* __restrict__ erbf,
                          const int* __restrict__ nbr,
                          const float* catW, const float* catB,
                          const float* edgW, const float* edgB,
                          float* __restrict__ mji){
  extern __shared__ __attribute__((aligned(64))) char smem[];
  h16*   Wc = (h16*)smem;                           // swz [192x64]
  float* bC = (float*)(smem + 192*64*2);            // 64
  float* We = bC + 64;                              // 6*64
  float* bE = We + 6*64;                            // 64
  h16*   XsAll = (h16*)(bE + 64);                   // 8 * 16*200
  __shared__ int I0s[8][16], I1s[8][16];
  __shared__ float Er[8][96];
  const int tid = threadIdx.x, nthr = blockDim.x;
  load_wswz(Wc, catW, 192, 192, 64, tid, nthr);
  load_f32(bC, catB, 64, tid, nthr);
  load_f32(We, edgW, 6*64, tid, nthr);
  load_f32(bE, edgB, 64, tid, nthr);
  __syncthreads();
  const int lane = tid & 31, wv = tid >> 5;
  const int row0 = (blockIdx.x*8 + wv) * 16;
  if(row0 >= NE) return;
  h16* Xs = XsAll + wv*(16*200);
  if(lane < 16){ int e = imin(row0 + lane, NE-1); I1s[wv][lane] = nbr[2*e+1]; I0s[wv][lane] = nbr[2*e]; }
  for(int i = lane; i < 96; i += 32){ int e = imin(row0 + i/6, NE-1); Er[wv][i] = erbf[(size_t)e*6 + i%6]; }
  wbar();
  for(int i = lane; i < 16*192; i += 32){
    int r = i / 192, c = i - r*192;
    float v;
    if(c < 64)       v = h[(size_t)I1s[wv][r]*64 + c];
    else if(c < 128) v = h[(size_t)I0s[wv][r]*64 + (c-64)];
    else {
      int ce = c - 128;
      float s = bE[ce];
#pragma unroll
      for(int q = 0; q < 6; ++q) s += Er[wv][r*6+q] * We[q*64 + ce];
      v = silu(s);
    }
    Xs[r*200 + c] = (h16)v;
  }
  wbar();
  v8f acc[4]; for(int n = 0; n < 4; ++n) acc[n] = vzero();
  gemm_swz<4>(Xs, 200, Wc, 192, acc, lane);
  store_acc_glb<4>(mji, row0, NE, acc, bC, true, lane);
}

// ---------------- fused edge readout ----------------------------------------
__global__ void k_edge_read(const float* __restrict__ erbf, const float* __restrict__ mji,
                            const float* einW,
                            const float* e0W, const float* e0B,
                            const float* e1W, const float* e1B,
                            const float* e2W, const float* e2B,
                            const float* e3W, const float* e3B,
                            float* __restrict__ ef, int accfl){
  extern __shared__ __attribute__((aligned(64))) char smem[];
  h16*   W0 = (h16*)smem;                     // swz [64x64]
  h16*   W1 = W0 + 64*64;                     // swz [64x64]
  h16*   W2 = W1 + 64*64;                     // swz [64x32]
  float* Wein = (float*)(W2 + 64*32);         // 6*64
  float* b0 = Wein + 6*64;                    // 64
  float* b1 = b0 + 64;                        // 64
  float* b2 = b1 + 64;                        // 32
  float* w3 = b2 + 32;                        // 32
  h16*   stg = (h16*)(w3 + 32);               // 8 * 2*16*72
  float* H2a = (float*)(stg + 8*2*16*72);     // 8 * 16*32
  __shared__ float Er[8][96];
  const int tid = threadIdx.x, nthr = blockDim.x;
  load_wswz(W0, e0W, 64, 64, 64, tid, nthr);  load_f32(b0, e0B, 64, tid, nthr);
  load_wswz(W1, e1W, 64, 64, 64, tid, nthr);  load_f32(b1, e1B, 64, tid, nthr);
  load_wswz(W2, e2W, 64, 64, 32, tid, nthr);  load_f32(b2, e2B, 32, tid, nthr);
  load_f32(Wein, einW, 6*64, tid, nthr);
  load_f32(w3, e3W, 32, tid, nthr);
  __syncthreads();
  const int lane = tid & 31, wv = tid >> 5;
  const int row0 = (blockIdx.x*8 + wv) * 16;
  if(row0 >= NE) return;
  h16* Xs = stg + wv*2*16*72;
  h16* Ys = Xs + 16*72;
  float* H2 = H2a + wv*16*32;
  for(int i = lane; i < 96; i += 32){ int e = imin(row0 + i/6, NE-1); Er[wv][i] = erbf[(size_t)e*6 + i%6]; }
  wbar();
  for(int i = lane; i < 16*64; i += 32){
    int r = i >> 6, c = i & 63;
    int e = imin(row0 + r, NE-1);
    float s = 0.0f;
#pragma unroll
    for(int q = 0; q < 6; ++q) s += Er[wv][r*6+q] * Wein[q*64 + c];
    Xs[r*72 + c] = (h16)(s * mji[(size_t)e*64 + c]);
  }
  wbar();
  v8f acc[4]; for(int n = 0; n < 4; ++n) acc[n] = vzero();
  gemm_swz<4>(Xs, 72, W0, 64, acc, lane);
  stage_acc_h16<4>(Ys, 72, acc, b0, true, lane); wbar();
  for(int n = 0; n < 4; ++n) acc[n] = vzero();
  gemm_swz<4>(Ys, 72, W1, 64, acc, lane);
  stage_acc_h16<4>(Xs, 72, acc, b1, true, lane); wbar();
  v8f a2[2]; a2[0] = vzero(); a2[1] = vzero();
  gemm_swz<2>(Xs, 72, W2, 64, a2, lane);
  { int colb = lane & 15, rb = (lane >> 4) * 8;
#pragma unroll
    for(int n = 0; n < 2; ++n)
#pragma unroll
      for(int v = 0; v < 8; ++v){ int c = n*16 + colb; H2[(rb+v)*32 + c] = silu(a2[n][v] + b2[c]); }
  }
  wbar();
  if(lane < 16){
    int e = row0 + lane;
    if(e < NE){
      float s = e3B[0];
#pragma unroll
      for(int k = 0; k < 32; ++k) s += H2[lane*32 + k] * w3[k];
      if(accfl) ef[e] += s; else ef[e] = s;
    }
  }
}

// ---------------- fused angle readout ---------------------------------------
__global__ void k_angle_read(const float* __restrict__ asbf, const float* __restrict__ mji,
                             const int* __restrict__ kj, const int* __restrict__ ji,
                             const float* ainW,
                             const float* a0W, const float* a0B,
                             const float* a1W, const float* a1B,
                             const float* a2W, const float* a2B,
                             const float* a3W, const float* a3B,
                             float* __restrict__ af, int accfl){
  extern __shared__ __attribute__((aligned(64))) char smem[];
  h16*   Wa = (h16*)smem;                     // swz [64x64] (rows>=42 zero)
  h16*   W0 = Wa + 64*64;                     // swz [64x64]
  h16*   W1 = W0 + 64*64;                     // swz [64x64]
  h16*   W2 = W1 + 64*64;                     // swz [64x32]
  float* b0 = (float*)(W2 + 64*32);           // 64
  float* b1 = b0 + 64;                        // 64
  float* b2 = b1 + 64;                        // 32
  float* w3 = b2 + 32;                        // 32
  h16*   stg = (h16*)(w3 + 32);               // 8 * 2*16*72
  float* Msa = (float*)(stg + 8*2*16*72);     // 8 * 16*64
  float* H2a = Msa + 8*16*64;                 // 8 * 16*32
  __shared__ int KJ[8][16], JI[8][16];
  const int tid = threadIdx.x, nthr = blockDim.x;
  load_wswz(Wa, ainW, NSBF, 64, 64, tid, nthr);
  load_wswz(W0, a0W, 64, 64, 64, tid, nthr);  load_f32(b0, a0B, 64, tid, nthr);
  load_wswz(W1, a1W, 64, 64, 64, tid, nthr);  load_f32(b1, a1B, 64, tid, nthr);
  load_wswz(W2, a2W, 64, 64, 32, tid, nthr);  load_f32(b2, a2B, 32, tid, nthr);
  load_f32(w3, a3W, 32, tid, nthr);
  __syncthreads();
  const int lane = tid & 31, wv = tid >> 5;
  const int row0 = (blockIdx.x*8 + wv) * 16;
  if(row0 >= NANG) return;
  h16* Xs = stg + wv*2*16*72;
  h16* Ys = Xs + 16*72;
  float* Ms = Msa + wv*16*64;
  float* H2 = H2a + wv*16*32;
  if(lane < 16){ int a = imin(row0 + lane, NANG-1); KJ[wv][lane] = kj[a]; JI[wv][lane] = ji[a]; }
  if(lane == 0 && row0 + 16 < NANG) __builtin_prefetch(asbf + (size_t)(row0+16)*NSBF, 0, 0);
  wbar();
  for(int i = lane; i < 16*64; i += 32){
    int r = i >> 6, c = i & 63;
    int a = imin(row0 + r, NANG-1);
    Xs[r*72 + c] = (h16)(c < NSBF ? asbf[(size_t)a*NSBF + c] : 0.0f);
    Ms[r*64 + c] = mji[(size_t)KJ[wv][r]*64 + c] + mji[(size_t)JI[wv][r]*64 + c];
  }
  wbar();
  v8f acc[4]; for(int n = 0; n < 4; ++n) acc[n] = vzero();
  gemm_swz<4>(Xs, 72, Wa, 64, acc, lane);
  { int colb = lane & 15, rb = (lane >> 4) * 8;     // af0 = (sbf@Wain) * (m_kj + m_ji)
#pragma unroll
    for(int n = 0; n < 4; ++n)
#pragma unroll
      for(int v = 0; v < 8; ++v){ int c = n*16 + colb, r = rb + v; Ys[r*72 + c] = (h16)(acc[n][v] * Ms[r*64 + c]); }
  }
  wbar();
  for(int n = 0; n < 4; ++n) acc[n] = vzero();
  gemm_swz<4>(Ys, 72, W0, 64, acc, lane);
  stage_acc_h16<4>(Xs, 72, acc, b0, true, lane); wbar();
  for(int n = 0; n < 4; ++n) acc[n] = vzero();
  gemm_swz<4>(Xs, 72, W1, 64, acc, lane);
  stage_acc_h16<4>(Ys, 72, acc, b1, true, lane); wbar();
  v8f a2[2]; a2[0] = vzero(); a2[1] = vzero();
  gemm_swz<2>(Ys, 72, W2, 64, a2, lane);
  { int colb = lane & 15, rb = (lane >> 4) * 8;
#pragma unroll
    for(int n = 0; n < 2; ++n)
#pragma unroll
      for(int v = 0; v < 8; ++v){ int c = n*16 + colb; H2[(rb+v)*32 + c] = silu(a2[n][v] + b2[c]); }
  }
  wbar();
  if(lane < 16){
    int a = row0 + lane;
    if(a < NANG){
      float s = a3B[0];
#pragma unroll
      for(int k = 0; k < 32; ++k) s += H2[lane*32 + k] * w3[k];
      if(accfl) af[a] += s; else af[a] = s;
    }
  }
}

// ---------------- interaction: x_ji, x_kj ------------------------------------
__global__ void k_int_xx(const float* __restrict__ mji, const float* __restrict__ erbf,
                         const float* jiW, const float* jiB,
                         const float* kjW, const float* kjB, const float* rbfW,
                         float* __restrict__ xji, float* __restrict__ xkj){
  extern __shared__ __attribute__((aligned(64))) char smem[];
  h16*   Wj = (h16*)smem;                     // swz [64x64]
  h16*   Wk = Wj + 64*64;                     // swz [64x64]
  float* bj = (float*)(Wk + 64*64);           // 64
  float* bk = bj + 64;                        // 64
  float* Wr = bk + 64;                        // 6*64
  h16*   stg = (h16*)(Wr + 6*64);             // 8 * 16*72
  __shared__ float Er[8][96];
  const int tid = threadIdx.x, nthr = blockDim.x;
  load_wswz(Wj, jiW, 64, 64, 64, tid, nthr);  load_f32(bj, jiB, 64, tid, nthr);
  load_wswz(Wk, kjW, 64, 64, 64, tid, nthr);  load_f32(bk, kjB, 64, tid, nthr);
  load_f32(Wr, rbfW, 6*64, tid, nthr);
  __syncthreads();
  const int lane = tid & 31, wv = tid >> 5;
  const int row0 = (blockIdx.x*8 + wv) * 16;
  if(row0 >= NE) return;
  h16* Xs = stg + wv*16*72;
  for(int i = lane; i < 96; i += 32){ int e = imin(row0 + i/6, NE-1); Er[wv][i] = erbf[(size_t)e*6 + i%6]; }
  for(int i = lane; i < 16*64; i += 32){
    int r = i >> 6, c = i & 63;
    int e = imin(row0 + r, NE-1);
    Xs[r*72 + c] = (h16)mji[(size_t)e*64 + c];
  }
  wbar();
  v8f acc[4]; for(int n = 0; n < 4; ++n) acc[n] = vzero();
  gemm_swz<4>(Xs, 72, Wj, 64, acc, lane);
  store_acc_glb<4>(xji, row0, NE, acc, bj, true, lane);
  for(int n = 0; n < 4; ++n) acc[n] = vzero();
  gemm_swz<4>(Xs, 72, Wk, 64, acc, lane);
  { int colb = lane & 15, rb = (lane >> 4) * 8;
#pragma unroll
    for(int n = 0; n < 4; ++n)
#pragma unroll
      for(int v = 0; v < 8; ++v){
        int r = rb + v, e = row0 + r;
        if(e < NE){
          int c = n*16 + colb;
          float rt = 0.0f;
#pragma unroll
          for(int q = 0; q < 6; ++q) rt += Er[wv][r*6+q] * Wr[q*64 + c];
          xkj[(size_t)e*64 + c] = silu(acc[n][v] + bk[c]) * rt;
        }
      }
  }
}

// ---------------- bilinear einsum as K=512 WMMA GEMM + segment-sum atomics ---
__global__ void k_bilinear(const float* __restrict__ asbf, const float* __restrict__ xkj,
                           const int* __restrict__ kj, const int* __restrict__ ji,
                           const float* sbfW, const float* bil,
                           float* __restrict__ agg){
  extern __shared__ __attribute__((aligned(64))) char smem[];
  h16*   Wb = (h16*)smem;                     // swz [512x64] (bil reshaped (d*8+b, e))
  float* sW = (float*)(Wb + 512*64);          // 42*8
  float* Xka = sW + NSBF*8;                   // 8 * 16*64
  float* Sva = Xka + 8*16*64;                 // 8 * 16*8
  __shared__ int KJ[8][16], JI[8][16];
  const int tid = threadIdx.x, nthr = blockDim.x;
  load_wswz(Wb, bil, 512, 512, 64, tid, nthr);
  load_f32(sW, sbfW, NSBF*8, tid, nthr);
  __syncthreads();
  const int lane = tid & 31, wv = tid >> 5;
  const int row0 = (blockIdx.x*8 + wv) * 16;
  if(row0 >= NANG) return;
  float* Xk = Xka + wv*16*64;
  float* Sv = Sva + wv*16*8;
  if(lane < 16){ int a = imin(row0 + lane, NANG-1); KJ[wv][lane] = kj[a]; JI[wv][lane] = ji[a]; }
  if(lane == 0 && row0 + 16 < NANG) __builtin_prefetch(asbf + (size_t)(row0+16)*NSBF, 0, 0);
  wbar();
  for(int i = lane; i < 16*64; i += 32){
    int r = i >> 6, c = i & 63;
    Xk[r*64 + c] = xkj[(size_t)KJ[wv][r]*64 + c];
  }
  for(int i = lane; i < 16*8; i += 32){
    int r = i >> 3, b = i & 7;
    int a = imin(row0 + r, NANG-1);
    float s = 0.0f;
    for(int q = 0; q < NSBF; ++q) s += asbf[(size_t)a*NSBF + q] * sW[q*8 + b];
    Sv[r*8 + b] = s;
  }
  wbar();
  v8f acc[4]; for(int n = 0; n < 4; ++n) acc[n] = vzero();
  const int arow = lane & 15;
  for(int kc = 0; kc < 512; kc += 32){
    const int kb = kc + ((lane >> 4) << 3);
    v16h a;
#pragma unroll
    for(int h = 0; h < 8; ++h){ int k = kb + h;       a[h]   = (h16)(Xk[arow*64 + (k>>3)] * Sv[arow*8 + (k&7)]); }
#pragma unroll
    for(int h = 0; h < 8; ++h){ int k = kb + 16 + h;  a[h+8] = (h16)(Xk[arow*64 + (k>>3)] * Sv[arow*8 + (k&7)]); }
#pragma unroll
    for(int n = 0; n < 4; ++n){
      v16h b = frag_b_swz(Wb, (kc >> 5)*4 + n, lane);
      acc[n] = __builtin_amdgcn_wmma_f32_16x16x32_f16(false, a, false, b, (short)0, acc[n], false, false);
    }
  }
  { int colb = lane & 15, rb = (lane >> 4) * 8;
#pragma unroll
    for(int n = 0; n < 4; ++n)
#pragma unroll
      for(int v = 0; v < 8; ++v){
        int ar = row0 + rb + v;
        if(ar < NANG){
          int e = JI[wv][rb + v];
          atomicAdd(&agg[(size_t)e*64 + n*16 + colb], acc[n][v]);
        }
      }
  }
}

// ---------------- residual update: m_ji += h + act(res1(act(res0(h)))) -------
__global__ void k_resid(const float* __restrict__ xji, const float* __restrict__ agg,
                        const float* r0W, const float* r0B,
                        const float* r1W, const float* r1B,
                        float* __restrict__ mji){
  extern __shared__ __attribute__((aligned(64))) char smem[];
  h16*   W0 = (h16*)smem;                     // swz [64x64]
  h16*   W1 = W0 + 64*64;                     // swz [64x64]
  float* b0 = (float*)(W1 + 64*64);           // 64
  float* b1 = b0 + 64;                        // 64
  h16*   stg = (h16*)(b1 + 64);               // 8 * 2*16*72
  float* Hfa = (float*)(stg + 8*2*16*72);     // 8 * 16*64
  const int tid = threadIdx.x, nthr = blockDim.x;
  load_wswz(W0, r0W, 64, 64, 64, tid, nthr);  load_f32(b0, r0B, 64, tid, nthr);
  load_wswz(W1, r1W, 64, 64, 64, tid, nthr);  load_f32(b1, r1B, 64, tid, nthr);
  __syncthreads();
  const int lane = tid & 31, wv = tid >> 5;
  const int row0 = (blockIdx.x*8 + wv) * 16;
  if(row0 >= NE) return;
  h16* Xs = stg + wv*2*16*72;
  h16* Ys = Xs + 16*72;
  float* Hf = Hfa + wv*16*64;
  for(int i = lane; i < 16*64; i += 32){
    int r = i >> 6, c = i & 63;
    int e = imin(row0 + r, NE-1);
    float hv = xji[(size_t)e*64 + c] + agg[(size_t)e*64 + c];
    Hf[r*64 + c] = hv;
    Xs[r*72 + c] = (h16)hv;
  }
  wbar();
  v8f acc[4]; for(int n = 0; n < 4; ++n) acc[n] = vzero();
  gemm_swz<4>(Xs, 72, W0, 64, acc, lane);
  stage_acc_h16<4>(Ys, 72, acc, b0, true, lane); wbar();
  for(int n = 0; n < 4; ++n) acc[n] = vzero();
  gemm_swz<4>(Ys, 72, W1, 64, acc, lane);
  { int colb = lane & 15, rb = (lane >> 4) * 8;
#pragma unroll
    for(int n = 0; n < 4; ++n)
#pragma unroll
      for(int v = 0; v < 8; ++v){
        int r = rb + v, e = row0 + r;
        if(e < NE){
          int c = n*16 + colb;
          float val = Hf[r*64 + c] + silu(acc[n][v] + b1[c]);
          mji[(size_t)e*64 + c] += val;
        }
      }
  }
}

// ---------------- forces -----------------------------------------------------
__global__ void k_edge_force(const float* __restrict__ nxyz, const int* __restrict__ nbr,
                             const float* __restrict__ ef, float* __restrict__ out){
  int e = blockIdx.x*blockDim.x + threadIdx.x;
  if(e >= NE) return;
  int i0 = nbr[2*e], i1 = nbr[2*e+1];
  float rx = nxyz[4*i0+1]-nxyz[4*i1+1];
  float ry = nxyz[4*i0+2]-nxyz[4*i1+2];
  float rz = nxyz[4*i0+3]-nxyz[4*i1+3];
  float dis = sqrtf(rx*rx + ry*ry + rz*rz + EPS3);
  float s = ef[e] / dis;
  atomicAdd(&out[3*i0+0],  s*rx); atomicAdd(&out[3*i0+1],  s*ry); atomicAdd(&out[3*i0+2],  s*rz);
  atomicAdd(&out[3*i1+0], -s*rx); atomicAdd(&out[3*i1+1], -s*ry); atomicAdd(&out[3*i1+2], -s*rz);
}

__global__ void k_angle_force(const float* __restrict__ nxyz, const int* __restrict__ ang,
                              const float* __restrict__ af, float* __restrict__ out){
  int a = blockIdx.x*blockDim.x + threadIdx.x;
  if(a >= NANG) return;
  int i = ang[3*a], j = ang[3*a+1], k = ang[3*a+2];
  float jx = nxyz[4*j+1], jy = nxyz[4*j+2], jz = nxyz[4*j+3];
  float ax = nxyz[4*i+1]-jx, ay = nxyz[4*i+2]-jy, az = nxyz[4*i+3]-jz;   // r_ji
  float bx = nxyz[4*k+1]-jx, by = nxyz[4*k+2]-jy, bz = nxyz[4*k+3]-jz;   // r_jk
  float d1 = sqrtf(ax*ax+ay*ay+az*az + EPS3);
  float d2 = sqrtf(bx*bx+by*by+bz*bz + EPS3);
  float i1 = 1.0f/d1, i2 = 1.0f/d2;
  float ux = bx*i2, uy = by*i2, uz = bz*i2;          // unit_jk
  float vx = ax*i1, vy = ay*i1, vz = az*i1;          // unit_ji
  float dau = ax*ux + ay*uy + az*uz;                 // r_ji . unit_jk
  float dbv = bx*vx + by*vy + bz*vz;                 // r_jk . unit_ji
  float q1 = i1*i1, q2 = i2*i2;
  // adj_ji = (-d1*u + r_ji*(r_ji.u)/d1)/d1^2 ; adj_jk symmetric
  float gx = af[a] * ((-d1*ux + ax*dau*i1)*q1 + (-d2*vx + bx*dbv*i2)*q2);
  float gy = af[a] * ((-d1*uy + ay*dau*i1)*q1 + (-d2*vy + by*dbv*i2)*q2);
  float gz = af[a] * ((-d1*uz + az*dau*i1)*q1 + (-d2*vz + bz*dbv*i2)*q2);
  atomicAdd(&out[3*j+0],  gx); atomicAdd(&out[3*j+1],  gy); atomicAdd(&out[3*j+2],  gz);
  atomicAdd(&out[3*i+0], -gx); atomicAdd(&out[3*i+1], -gy); atomicAdd(&out[3*i+2], -gz);
}

// ---------------- host -------------------------------------------------------
namespace {
constexpr int SM_EMBED = 192*64*2 + 64*4 + 6*64*4 + 64*4 + 8*16*200*2;                 // 77824
constexpr int SM_EREAD = (64*64 + 64*64 + 64*32)*2 + (6*64 + 64 + 64 + 32 + 32)*4
                       + 8*2*16*72*2 + 8*16*32*4;                                       // 76032
constexpr int SM_AREAD = (64*64*3 + 64*32)*2 + (64 + 64 + 32 + 32)*4
                       + 8*2*16*72*2 + 8*16*64*4 + 8*16*32*4;                           // 115456
constexpr int SM_INT   = (64*64 + 64*64)*2 + (64 + 64 + 6*64)*4 + 8*16*72*2;            // 36864
constexpr int SM_BIL   = 512*64*2 + NSBF*8*4 + 8*16*64*4 + 8*16*8*4;                    // 103744
constexpr int SM_RES   = (64*64 + 64*64)*2 + (64 + 64)*4 + 8*2*16*72*2 + 8*16*64*4;     // 86528
}

extern "C" void kernel_launch(void* const* d_in, const int* in_sizes, int n_in,
                              void* d_out, int out_size, void* d_ws, size_t ws_size,
                              hipStream_t stream){
  (void)in_sizes; (void)n_in; (void)out_size; (void)ws_size;
  const float* nxyz = (const float*)d_in[0];
  const int*   nbr  = (const int*)d_in[1];
  const int*   ang  = (const int*)d_in[2];
  const int*   kj   = (const int*)d_in[3];
  const int*   ji   = (const int*)d_in[4];
  // params flattened in jax tree order (sorted dict keys at each level)
  auto P = [&](int i)->const float*{ return (const float*)d_in[5 + i]; };
  const float *catW = P(0), *catB = P(1), *edgW = P(2), *edgB = P(3), *table = P(4);
  struct Conv { const float *bil,*jiW,*jiB,*kjW,*kjB,*rbfW,*r0W,*r0B,*r1W,*r1B,*sbfW; } cv[3];
  for(int c = 0; c < 3; ++c){
    int b = 5 + 11*c;
    cv[c] = { P(b+0),P(b+1),P(b+2),P(b+3),P(b+4),P(b+5),P(b+6),P(b+7),P(b+8),P(b+9),P(b+10) };
  }
  struct Read { const float *a0W,*a0B,*a1W,*a1B,*a2W,*a2B,*a3W,*a3B,*ainW,
                             *e0W,*e0B,*e1W,*e1B,*e2W,*e2B,*e3W,*e3B,*einW; } rd[4];
  for(int r = 0; r < 4; ++r){
    int b = 38 + 18*r;
    rd[r] = { P(b+0),P(b+1),P(b+2),P(b+3),P(b+4),P(b+5),P(b+6),P(b+7),P(b+8),
              P(b+9),P(b+10),P(b+11),P(b+12),P(b+13),P(b+14),P(b+15),P(b+16),P(b+17) };
  }
  // workspace partition (floats)
  float* W = (float*)d_ws;
  size_t o = 0;
  float* d_e  = W + o; o += NE;
  float* erbf = W + o; o += (size_t)NE*6;
  float* asbf = W + o; o += (size_t)NANG*NSBF;
  float* hA   = W + o; o += (size_t)NA*64;
  float* mji  = W + o; o += (size_t)NE*64;
  float* xji  = W + o; o += (size_t)NE*64;
  float* xkj  = W + o; o += (size_t)NE*64;
  float* agg  = W + o; o += (size_t)NE*64;
  float* efb  = W + o; o += NE;
  float* afb  = W + o; o += NANG;
  float* out  = (float*)d_out;

  auto cdiv = [](long long a, long long b)->int{ return (int)((a + b - 1) / b); };
  const int eB = cdiv(cdiv(NE, 16), 8);     // 8 waves/block, 16 rows/wave
  const int aB = cdiv(cdiv(NANG, 16), 8);

  k_zero<<<cdiv(NA*3, 256), 256, 0, stream>>>(out, (long long)NA*3);
  k_edge_geom<<<cdiv(NE, 256), 256, 0, stream>>>(nxyz, nbr, d_e, erbf);
  k_angle_sbf<<<cdiv(NANG, 256), 256, 0, stream>>>(nxyz, ang, kj, d_e, asbf);
  k_embed_h<<<cdiv((long long)NA*64, 256), 256, 0, stream>>>(nxyz, table, hA);
  k_embed_m<<<eB, 256, SM_EMBED, stream>>>(hA, erbf, nbr, catW, catB, edgW, edgB, mji);

  k_edge_read<<<eB, 256, SM_EREAD, stream>>>(erbf, mji, rd[0].einW,
      rd[0].e0W, rd[0].e0B, rd[0].e1W, rd[0].e1B, rd[0].e2W, rd[0].e2B,
      rd[0].e3W, rd[0].e3B, efb, 0);
  k_angle_read<<<aB, 256, SM_AREAD, stream>>>(asbf, mji, kj, ji, rd[0].ainW,
      rd[0].a0W, rd[0].a0B, rd[0].a1W, rd[0].a1B, rd[0].a2W, rd[0].a2B,
      rd[0].a3W, rd[0].a3B, afb, 0);

  for(int c = 0; c < 3; ++c){
    k_int_xx<<<eB, 256, SM_INT, stream>>>(mji, erbf, cv[c].jiW, cv[c].jiB,
        cv[c].kjW, cv[c].kjB, cv[c].rbfW, xji, xkj);
    k_zero<<<cdiv((long long)NE*64, 256), 256, 0, stream>>>(agg, (long long)NE*64);
    k_bilinear<<<aB, 256, SM_BIL, stream>>>(asbf, xkj, kj, ji, cv[c].sbfW, cv[c].bil, agg);
    k_resid<<<eB, 256, SM_RES, stream>>>(xji, agg, cv[c].r0W, cv[c].r0B,
        cv[c].r1W, cv[c].r1B, mji);
    k_edge_read<<<eB, 256, SM_EREAD, stream>>>(erbf, mji, rd[c+1].einW,
        rd[c+1].e0W, rd[c+1].e0B, rd[c+1].e1W, rd[c+1].e1B, rd[c+1].e2W, rd[c+1].e2B,
        rd[c+1].e3W, rd[c+1].e3B, efb, 1);
    k_angle_read<<<aB, 256, SM_AREAD, stream>>>(asbf, mji, kj, ji, rd[c+1].ainW,
        rd[c+1].a0W, rd[c+1].a0B, rd[c+1].a1W, rd[c+1].a1B, rd[c+1].a2W, rd[c+1].a2B,
        rd[c+1].a3W, rd[c+1].a3B, afb, 1);
  }

  k_edge_force<<<cdiv(NE, 256), 256, 0, stream>>>(nxyz, nbr, efb, out);
  k_angle_force<<<cdiv(NANG, 256), 256, 0, stream>>>(nxyz, ang, afb, out);
}
